// EdgeLearner_85899346395
// MI455X (gfx1250) — matrix operand order repaired and checked by reading, not compiled, not run on hardware
//
#include <hip/hip_runtime.h>

#define Bc 4
#define Vc 64
#define Lc 1024
#define Dc 256
#define Ec 512

typedef __bf16 bf16;
typedef __attribute__((ext_vector_type(16))) __bf16 v16bf;
typedef __attribute__((ext_vector_type(8)))  __bf16 v8bf;
typedef __attribute__((ext_vector_type(8)))  float  v8f;
typedef __attribute__((ext_vector_type(4))) unsigned int u32x4;
typedef __attribute__((ext_vector_type(8))) int          i32x8;
typedef __attribute__((ext_vector_type(4))) int          i32x4;

#if defined(__has_builtin)
#if __has_builtin(__builtin_amdgcn_tensor_load_to_lds) && __has_builtin(__builtin_amdgcn_s_wait_tensorcnt)
#define HAVE_TDM 1
#endif
#endif

// ---------------- weight pre-conversion (runs once per launch, tiny) ----------------
__global__ void cvt_weights(const float* __restrict__ Wq, const float* __restrict__ Wk,
                            bf16* __restrict__ WqB, bf16* __restrict__ WkB) {
  int i = blockIdx.x * blockDim.x + threadIdx.x;
  if (i < Dc * Dc) {
    WqB[i] = (bf16)Wq[i];
    WkB[i] = (bf16)Wk[i];
  }
}

// A fragment (16x32, MxK) of row-major X (stride ld, elements).
__device__ __forceinline__ v16bf load_a_frag(const bf16* X, int ld, int row0, int k0, int lane) {
  const int row = row0 + (lane & 15);
  const int kb  = (lane >> 4) * 16;
  const bf16* p = X + row * ld + k0 + kb;
  v8bf lo = *(const v8bf*)p;          // 16B contiguous
  v8bf hi = *(const v8bf*)(p + 8);    // 16B contiguous
  v16bf a;
#pragma unroll
  for (int i = 0; i < 8; ++i) { a[i] = lo[i]; a[8 + i] = hi[i]; }
  return a;
}

// B fragment (32x16, KxN); source Yt stores B transposed: Yt[n][k] k-contiguous.
__device__ __forceinline__ v16bf load_b_frag(const bf16* Yt, int ld, int col0, int k0, int lane) {
  const int n  = col0 + (lane & 15);
  const int kb = (lane >> 4) * 16;
  const bf16* p = Yt + n * ld + k0 + kb;
  return *(const v16bf*)p;            // 32B contiguous
}

// ---------------- fused EdgeLearner: one workgroup per (b, l) ----------------
__launch_bounds__(256, 1)
__global__ void edge_learner_fused(const float* __restrict__ hs,
                                   const long long* __restrict__ edge_index,
                                   const float* __restrict__ ew,
                                   const bf16* __restrict__ WqB,
                                   const float* __restrict__ bq,
                                   const bf16* __restrict__ WkB,
                                   const float* __restrict__ bk,
                                   float* __restrict__ out) {
  __shared__ __align__(16) float stage[Vc][Dc];   // 64 KB: TDM dest, later S+base
  __shared__ __align__(16) bf16  hsb[Vc][Dc];     // 32 KB
  __shared__ __align__(16) bf16  Qb[Vc][Dc];      // 32 KB
  __shared__ __align__(16) bf16  Kb[Vc][Dc];      // 32 KB
  float (*S)[Vc]    = (float (*)[Vc])&stage[0][0];    // 16 KB (rows 0..15)
  float (*base)[Vc] = (float (*)[Vc])&stage[16][0];   // 16 KB (rows 16..31)

  const int blk  = blockIdx.x;
  const int b    = blk >> 10;       // / L
  const int l    = blk & (Lc - 1);
  const int t    = threadIdx.x;
  const int lane = t & 31;          // wave32
  const int wv   = t >> 5;          // 8 waves

#ifdef HAVE_TDM
  // ---- TDM: each wave DMAs its 8 rows hs[b, v0:v0+8, l, 0:256] (f32) into LDS ----
  {
    const int v0 = wv * 8;
    const unsigned long long ga =
        (unsigned long long)(const void*)(hs + (((size_t)(b * Vc + v0) * Lc + l) * Dc));
    const unsigned ldsa = (unsigned)(unsigned long long)(const void*)&stage[v0][0];
    u32x4 g0;
    g0[0] = 1u;                                                  // count=1 (valid D#)
    g0[1] = ldsa;                                                // lds_addr
    g0[2] = (unsigned)ga;                                        // global_addr[31:0]
    g0[3] = (unsigned)((ga >> 32) & 0x01FFFFFFu) | 0x80000000u;  // addr[56:32] | type=2
    i32x8 g1;
    g1[0] = 0x20000;             // workgroup_mask=0, data_size=2 (4 bytes)
    g1[1] = (int)(256u << 16);   // tensor_dim0 = 256 (bits 63:48)
    g1[2] = (int)(64u << 16);    // tensor_dim1 = 64  (bits 95:80)
    g1[3] = (int)(256u << 16);   // tile_dim0 = 256   (bits 127:112)
    g1[4] = 8;                   // tile_dim1 = 8, tile_dim2 = 0
    g1[5] = (int)(Lc * Dc);      // tensor_dim0_stride = 262144 (bits 191:160)
    g1[6] = 0;
    g1[7] = 0;
    i32x4 gz = {0, 0, 0, 0};     // groups 2/3 unused (2D tile)
#if __clang_major__ >= 23
    i32x8 gz8 = {0, 0, 0, 0, 0, 0, 0, 0};
    __builtin_amdgcn_tensor_load_to_lds(g0, g1, gz, gz, gz8, 0);
#else
    __builtin_amdgcn_tensor_load_to_lds(g0, g1, gz, gz, 0);
#endif
    __builtin_amdgcn_s_wait_tensorcnt(0);
  }
  __syncthreads();

  // ---- LDS f32 -> LDS bf16 repack (each thread: one 256B chunk) ----
  {
    const int v  = t >> 2;
    const int d0 = (t & 3) * 64;
#pragma unroll
    for (int i = 0; i < 64; i += 4) {
      float4 f = *(const float4*)&stage[v][d0 + i];
      hsb[v][d0 + i + 0] = (bf16)f.x;
      hsb[v][d0 + i + 1] = (bf16)f.y;
      hsb[v][d0 + i + 2] = (bf16)f.z;
      hsb[v][d0 + i + 3] = (bf16)f.w;
    }
    if (l + 1 < Lc)
      __builtin_prefetch(hs + (((size_t)(b * Vc + v) * Lc + l + 1) * Dc + d0), 0, 1);
  }
#else
  // ---- fallback staging: global f32 -> regs -> bf16 LDS ----
  {
    const int v  = t >> 2;
    const int d0 = (t & 3) * 64;
    const float* src = hs + (((size_t)(b * Vc + v) * Lc + l) * Dc + d0);
#pragma unroll
    for (int i = 0; i < 64; i += 4) {
      float4 f = *(const float4*)(src + i);
      hsb[v][d0 + i + 0] = (bf16)f.x;
      hsb[v][d0 + i + 1] = (bf16)f.y;
      hsb[v][d0 + i + 2] = (bf16)f.z;
      hsb[v][d0 + i + 3] = (bf16)f.w;
    }
    if (l + 1 < Lc) __builtin_prefetch(src + Dc, 0, 1);
  }
#endif
  __syncthreads();

  // ---- GEMM1: Q = hsb @ Wq^T + bq, K = hsb @ Wk^T + bk (128 16x16 tiles, 16/wave) ----
#pragma unroll
  for (int it = 0; it < 16; ++it) {
    const int  tid  = wv * 16 + it;       // 0..127
    const bool isK  = tid >= 64;
    const int  qt   = tid & 63;
    const int  mt   = qt >> 4;            // 0..3  (v tile)
    const int  nt   = qt & 15;            // 0..15 (e tile)
    const bf16*  W    = isK ? WkB : WqB;  // row-major (out,in): already "B transposed"
    const float* bias = isK ? bk  : bq;
    v8f acc = {};
#pragma unroll
    for (int k0 = 0; k0 < Dc; k0 += 32) {
      v16bf a  = load_a_frag(&hsb[0][0], Dc, mt * 16, k0, lane);
      v16bf bb = load_b_frag(W,          Dc, nt * 16, k0, lane);
      acc = __builtin_amdgcn_wmma_f32_16x16x32_bf16(false, a, false, bb,
                                                    (short)0, acc, false, false);
    }
    const int cc = lane & 15;
    const int rh = (lane >> 4) * 8;
    const float bv = bias[nt * 16 + cc];
    bf16 (*Dst)[Dc] = isK ? Kb : Qb;
#pragma unroll
    for (int i = 0; i < 8; ++i)
      Dst[mt * 16 + rh + i][nt * 16 + cc] = (bf16)(acc[i] + bv);
  }
  __syncthreads();   // stage f32 tile dead; region becomes S + base

  // ---- zero base adjacency ----
  for (int n = t; n < Vc * Vc; n += 256) base[n >> 6][n & 63] = 0.0f;
  __syncthreads();

  // ---- scatter edges of this batch into base (to_dense_adj sums duplicates) ----
  for (int e = t; e < Ec; e += 256) {
    const int ei = b * Ec + e;
    const int r  = (int)(edge_index[ei] % Vc);            // row 0: src
    const int c  = (int)(edge_index[Bc * Ec + ei] % Vc);  // row 1: dst
    atomicAdd(&base[r][c], ew[ei]);
  }

  // ---- GEMM2: S = (Q @ K^T) / sqrt(D)  (16 tiles, 2/wave) ----
#pragma unroll
  for (int it = 0; it < 2; ++it) {
    const int tid = wv * 2 + it;          // 0..15
    const int mt  = tid >> 2;
    const int nt  = tid & 3;
    v8f acc = {};
#pragma unroll
    for (int k0 = 0; k0 < Dc; k0 += 32) {
      v16bf a  = load_a_frag(&Qb[0][0], Dc, mt * 16, k0, lane);
      v16bf bb = load_b_frag(&Kb[0][0], Dc, nt * 16, k0, lane);
      acc = __builtin_amdgcn_wmma_f32_16x16x32_bf16(false, a, false, bb,
                                                    (short)0, acc, false, false);
    }
    const int cc = lane & 15;
    const int rh = (lane >> 4) * 8;
#pragma unroll
    for (int i = 0; i < 8; ++i)
      S[mt * 16 + rh + i][nt * 16 + cc] = acc[i] * 0.0625f;  // 1/sqrt(256)
  }
  __syncthreads();

  // ---- softmax over w (axis=2), temperature 0.01 ----
  if (t < Vc) {
    const int v = t;
    float m = -3.4e38f;
    for (int w = 0; w < Vc; ++w) m = fmaxf(m, S[v][w]);
    float sum = 0.f;
    for (int w = 0; w < Vc; ++w) { float x = __expf((S[v][w] - m) * 100.0f); S[v][w] = x; sum += x; }
    const float inv = 1.0f / sum;
    for (int w = 0; w < Vc; ++w) S[v][w] *= inv;
  }
  __syncthreads();

  // ---- softmax over v (axis=1), temperature 0.01 ----
  if (t < Vc) {
    const int w = t;
    float m = -3.4e38f;
    for (int v = 0; v < Vc; ++v) m = fmaxf(m, S[v][w]);
    float sum = 0.f;
    for (int v = 0; v < Vc; ++v) { float x = __expf((S[v][w] - m) * 100.0f); S[v][w] = x; sum += x; }
    const float inv = 1.0f / sum;
    for (int v = 0; v < Vc; ++v) S[v][w] *= inv;
  }
  __syncthreads();

  // ---- symmetrize, threshold, + base, diag := 1, write adj[b,v,w,l] ----
  const size_t obase = ((size_t)b * Vc * Vc) << 10;   // * L
  for (int n = t; n < Vc * Vc; n += 256) {
    const int v = n >> 6, w = n & 63;
    float a = 0.5f * (S[v][w] + S[w][v]);
    a = (a >= 0.1f) ? a : 0.0f;
    a += base[v][w];
    if (v == w) a = 1.0f;
    out[obase + (((size_t)n) << 10) + l] = a;
  }
}

extern "C" void kernel_launch(void* const* d_in, const int* in_sizes, int n_in,
                              void* d_out, int out_size, void* d_ws, size_t ws_size,
                              hipStream_t stream) {
  (void)in_sizes; (void)n_in; (void)out_size; (void)ws_size;
  const float*     hs = (const float*)d_in[0];
  const long long* ei = (const long long*)d_in[1];   // int64 edge_index
  const float*     ew = (const float*)d_in[2];
  const float*     Wq = (const float*)d_in[3];
  const float*     bq = (const float*)d_in[4];
  const float*     Wk = (const float*)d_in[5];
  const float*     bk = (const float*)d_in[6];
  float* out = (float*)d_out;

  bf16* WqB = (bf16*)d_ws;            // 128 KB
  bf16* WkB = WqB + Dc * Dc;          // 128 KB

  cvt_weights<<<(Dc * Dc + 255) / 256, 256, 0, stream>>>(Wq, Wk, WqB, WkB);
  edge_learner_fused<<<Bc * Lc, 256, 0, stream>>>(hs, ei, ew, WqB, bq, WkB, bk, out);
}
